// GNSModel_50706383897185
// MI455X (gfx1250) — matrix operand order repaired and checked by reading, not compiled, not run on hardware
//
#include <hip/hip_runtime.h>
#include <hip/hip_fp16.h>

// ---------------------------------------------------------------------------
// GNS-style message-passing GNN on gfx1250 using V_WMMA_F32_16X16X32_F16.
// All GEMMs: f16 A/B, f32 accumulate. Weights pre-swizzled to B-fragment
// layout. Hidden activations relaid out through LDS (ds ops). Aggregation via
// global f32 atomics.
// ---------------------------------------------------------------------------

#define NN 20000      // nodes
#define NE 320000     // edges
#define LDIM 128
#define NLAYERS 5

typedef _Float16 v16h __attribute__((ext_vector_type(16)));
typedef _Float16 v8h  __attribute__((ext_vector_type(8)));
typedef float    v8f  __attribute__((ext_vector_type(8)));
typedef float    v4f  __attribute__((ext_vector_type(4)));

__device__ __forceinline__ v8f wmma_f16(v16h a, v16h b, v8f c) {
  // D = A(16x32 f16) * B(32x16 f16) + C(16x16 f32)
  return __builtin_amdgcn_wmma_f32_16x16x32_f16(
      /*neg_a=*/false, a, /*neg_b=*/false, b,
      /*c_mod=*/(short)0, c, /*reuse_a=*/false, /*reuse_b=*/false);
}

__device__ __forceinline__ v16h cat8(v8h lo, v8h hi) {
  return __builtin_shufflevector(lo, hi, 0, 1, 2, 3, 4, 5, 6, 7,
                                 8, 9, 10, 11, 12, 13, 14, 15);
}

// A fragment (16x32, f16) from a row-major f16 row. Per ISA layout:
// lane m=lane&15 owns row m; hsel=lane>>4. Element e of the v16h:
//   K = ktOff + (e/8)*16 + hsel*8 + (e%8)
// -> two contiguous 16B loads at ktOff+hsel*8 and ktOff+16+hsel*8.
__device__ __forceinline__ v16h load_a_f16(const _Float16* row, int ktOff, int hsel) {
  const _Float16* p = row + ktOff + hsel * 8;
  v8h lo = *(const v8h*)(p);
  v8h hi = *(const v8h*)(p + 16);
  return cat8(lo, hi);
}

// Same fragment but sourced from a row-major f32 row (converted to f16).
__device__ __forceinline__ v16h load_a_f32(const float* row, int ktOff, int hsel) {
  const float* p = row + ktOff + hsel * 8;
  v4f a0 = *(const v4f*)(p);
  v4f a1 = *(const v4f*)(p + 4);
  v4f b0 = *(const v4f*)(p + 16);
  v4f b1 = *(const v4f*)(p + 20);
  v16h r;
#pragma unroll
  for (int i = 0; i < 4; ++i) {
    r[i]      = (_Float16)a0[i];
    r[4 + i]  = (_Float16)a1[i];
    r[8 + i]  = (_Float16)b0[i];
    r[12 + i] = (_Float16)b1[i];
  }
  return r;
}

// B fragment from pre-swizzled weights: layout [kt][nt][lane][16 halves],
// i.e. one contiguous 32B load per lane.
__device__ __forceinline__ v16h load_b(const _Float16* w, int kt, int nt, int NT, int lane) {
  return *(const v16h*)(w + ((size_t)((kt * NT + nt) * 32 + lane) << 4));
}

__device__ __forceinline__ void atom_add_f32(float* p, float v) {
  unsafeAtomicAdd(p, v);  // -> global_atomic_add_f32
}

// ---------------------------------------------------------------------------
// Prep kernels
// ---------------------------------------------------------------------------

// Convert f32 weight [Ksrc,Nsrc] (row-major) -> f16 fragments [Kpad/32][Npad/16][32][16]
__global__ void swizzle_weight(const float* __restrict__ src, _Float16* __restrict__ dst,
                               int Ksrc, int Nsrc, int Kpad, int Npad) {
  int total = (Kpad / 32) * (Npad / 16) * 512;
  int t = blockIdx.x * blockDim.x + threadIdx.x;
  if (t >= total) return;
  int j = t & 15;
  int lane = (t >> 4) & 31;
  int rest = t >> 9;                // kt*NT + nt
  int NT = Npad / 16;
  int nt = rest % NT;
  int kt = rest / NT;
  int K = kt * 32 + (j >> 3) * 16 + (lane >> 4) * 8 + (j & 7);
  int col = nt * 16 + (lane & 15);
  float v = (K < Ksrc && col < Nsrc) ? src[(size_t)K * Nsrc + col] : 0.0f;
  dst[t] = (_Float16)v;
}

// Pad/convert f32 [rows,scols] -> f16 [rows,dcols] with zero-padding.
__global__ void pad_f16(const float* __restrict__ src, _Float16* __restrict__ dst,
                        int rows, int scols, int dcols) {
  int t = blockIdx.x * blockDim.x + threadIdx.x;
  int total = rows * dcols;
  if (t >= total) return;
  int r = t / dcols, c = t - r * dcols;
  dst[t] = (_Float16)((c < scols) ? src[(size_t)r * scols + c] : 0.0f);
}

__global__ void zero_f32(float* __restrict__ p, int n) {
  int t = blockIdx.x * blockDim.x + threadIdx.x;
  if (t < n) p[t] = 0.0f;
}

// ---------------------------------------------------------------------------
// Encoder: rows x 32 -> 128 -> PReLU -> 128. One wave per 16-row tile.
// ---------------------------------------------------------------------------
template <bool WRITE32>
__global__ __launch_bounds__(256) void encoder_kernel(
    const _Float16* __restrict__ in16,   // [rows,32] f16, zero-padded
    const _Float16* __restrict__ w1,     // fragments, 1 kt x 8 nt
    const float* __restrict__ b1, const float* __restrict__ aPtr,
    const _Float16* __restrict__ w2,     // fragments, 4 kt x 8 nt
    const float* __restrict__ b2,
    _Float16* __restrict__ out16, float* __restrict__ out32, int ntiles) {
  __shared__ __align__(16) _Float16 sh[8][2048];   // per-wave 16x128 hidden tile
  const int wave = threadIdx.x >> 5, lane = threadIdx.x & 31;
  const int tile = blockIdx.x * 8 + wave;
  if (tile >= ntiles) return;                       // wave-uniform
  const int m = lane & 15, hsel = lane >> 4;
  const int row = tile * 16 + m;
  const float alpha = aPtr[0];

  v16h A0 = load_a_f16(in16 + (size_t)row * 32, 0, hsel);

#pragma unroll
  for (int nt = 0; nt < 8; ++nt) {
    v8f c = {};
    c = wmma_f16(A0, load_b(w1, 0, nt, 8, lane), c);
    const int col = nt * 16 + m;
    const float bb = b1[col];
#pragma unroll
    for (int v = 0; v < 8; ++v) {
      float hv = c[v] + bb;
      hv = hv > 0.0f ? hv : alpha * hv;
      sh[wave][(hsel * 8 + v) * 128 + col] = (_Float16)hv;
    }
  }

  v8f c2[8];
#pragma unroll
  for (int nt = 0; nt < 8; ++nt) c2[nt] = (v8f){};
  const _Float16* lr = &sh[wave][m * 128];
#pragma unroll
  for (int kt = 0; kt < 4; ++kt) {
    v16h a2 = load_a_f16(lr, kt * 32, hsel);
#pragma unroll
    for (int nt = 0; nt < 8; ++nt)
      c2[nt] = wmma_f16(a2, load_b(w2, kt, nt, 8, lane), c2[nt]);
  }

  const int base = tile * 16;
#pragma unroll
  for (int nt = 0; nt < 8; ++nt) {
    const int col = nt * 16 + m;
    const float bb = b2[col];
#pragma unroll
    for (int v = 0; v < 8; ++v) {
      const int rr = base + hsel * 8 + v;
      const float val = c2[nt][v] + bb;
      out16[(size_t)rr * 128 + col] = (_Float16)val;
      if (WRITE32) out32[(size_t)rr * 128 + col] = val;
    }
  }
}

// ---------------------------------------------------------------------------
// Edge message: m = MLP([h[dst], h[src], e]) ; agg[dst] += m   (per layer)
// One wave per 16-edge tile. GEMM1: 12 kt x 8 nt. GEMM2: 4 kt x 8 nt.
// ---------------------------------------------------------------------------
__global__ __launch_bounds__(256) void edge_msg_kernel(
    const _Float16* __restrict__ h16, const _Float16* __restrict__ e16,
    const int* __restrict__ eidx,    // [2,E]: row0=src, row1=dst
    const _Float16* __restrict__ w1, const float* __restrict__ b1,
    const float* __restrict__ aPtr,
    const _Float16* __restrict__ w2, const float* __restrict__ b2,
    float* __restrict__ agg, int E) {
  __shared__ __align__(16) _Float16 sh[8][2048];
  __shared__ int dsti[8][16];
  const int wave = threadIdx.x >> 5, lane = threadIdx.x & 31;
  const int tile = blockIdx.x * 8 + wave;
  const int m = lane & 15, hsel = lane >> 4;
  const int e = tile * 16 + m;      // E is a multiple of 16*8, always in range
  const int srow = eidx[e];
  const int drow = eidx[E + e];
  if (lane < 16) dsti[wave][lane] = drow;
  const float alpha = aPtr[0];

  // Gather A fragments: [h[dst] | h[src] | e]  (12 k-tiles of 32)
  v16h A[12];
  const _Float16* hd = h16 + (size_t)drow * 128;
  const _Float16* hs = h16 + (size_t)srow * 128;
  const _Float16* ep = e16 + (size_t)e * 128;
#pragma unroll
  for (int kt = 0; kt < 4; ++kt) {
    A[kt]     = load_a_f16(hd, kt * 32, hsel);
    A[4 + kt] = load_a_f16(hs, kt * 32, hsel);
    A[8 + kt] = load_a_f16(ep, kt * 32, hsel);
  }

  // GEMM1 + bias + PReLU -> LDS hidden tile
#pragma unroll
  for (int nt = 0; nt < 8; ++nt) {
    v8f c = {};
#pragma unroll
    for (int kt = 0; kt < 12; ++kt)
      c = wmma_f16(A[kt], load_b(w1, kt, nt, 8, lane), c);
    const int col = nt * 16 + m;
    const float bb = b1[col];
#pragma unroll
    for (int v = 0; v < 8; ++v) {
      float hv = c[v] + bb;
      hv = hv > 0.0f ? hv : alpha * hv;
      sh[wave][(hsel * 8 + v) * 128 + col] = (_Float16)hv;
    }
  }

  // GEMM2
  v8f c2[8];
#pragma unroll
  for (int nt = 0; nt < 8; ++nt) c2[nt] = (v8f){};
  const _Float16* lr = &sh[wave][m * 128];
#pragma unroll
  for (int kt = 0; kt < 4; ++kt) {
    v16h a2 = load_a_f16(lr, kt * 32, hsel);
#pragma unroll
    for (int nt = 0; nt < 8; ++nt)
      c2[nt] = wmma_f16(a2, load_b(w2, kt, nt, 8, lane), c2[nt]);
  }

  // Scatter-add (bias per edge message included, as in reference)
#pragma unroll
  for (int nt = 0; nt < 8; ++nt) {
    const int col = nt * 16 + m;
    const float bb = b2[col];
#pragma unroll
    for (int v = 0; v < 8; ++v) {
      const int r = dsti[wave][hsel * 8 + v];
      atom_add_f32(&agg[(size_t)r * 128 + col], c2[nt][v] + bb);
    }
  }
}

// ---------------------------------------------------------------------------
// Node update: h += MLP([h, agg])   (per layer). 8 kt x 8 nt then 4 kt x 8 nt.
// ---------------------------------------------------------------------------
__global__ __launch_bounds__(256) void node_upd_kernel(
    _Float16* __restrict__ h16, float* __restrict__ h32,
    const float* __restrict__ agg,
    const _Float16* __restrict__ w1, const float* __restrict__ b1,
    const float* __restrict__ aPtr,
    const _Float16* __restrict__ w2, const float* __restrict__ b2, int ntiles) {
  __shared__ __align__(16) _Float16 sh[8][2048];
  const int wave = threadIdx.x >> 5, lane = threadIdx.x & 31;
  const int tile = blockIdx.x * 8 + wave;
  if (tile >= ntiles) return;
  const int m = lane & 15, hsel = lane >> 4;
  const int base = tile * 16;
  const int row = base + m;
  const float alpha = aPtr[0];

  v16h A[8];
  const _Float16* hr = h16 + (size_t)row * 128;
  const float* ar = agg + (size_t)row * 128;
#pragma unroll
  for (int kt = 0; kt < 4; ++kt) {
    A[kt]     = load_a_f16(hr, kt * 32, hsel);
    A[4 + kt] = load_a_f32(ar, kt * 32, hsel);
  }

#pragma unroll
  for (int nt = 0; nt < 8; ++nt) {
    v8f c = {};
#pragma unroll
    for (int kt = 0; kt < 8; ++kt)
      c = wmma_f16(A[kt], load_b(w1, kt, nt, 8, lane), c);
    const int col = nt * 16 + m;
    const float bb = b1[col];
#pragma unroll
    for (int v = 0; v < 8; ++v) {
      float hv = c[v] + bb;
      hv = hv > 0.0f ? hv : alpha * hv;
      sh[wave][(hsel * 8 + v) * 128 + col] = (_Float16)hv;
    }
  }

  v8f c2[8];
#pragma unroll
  for (int nt = 0; nt < 8; ++nt) c2[nt] = (v8f){};
  const _Float16* lr = &sh[wave][m * 128];
#pragma unroll
  for (int kt = 0; kt < 4; ++kt) {
    v16h a2 = load_a_f16(lr, kt * 32, hsel);
#pragma unroll
    for (int nt = 0; nt < 8; ++nt)
      c2[nt] = wmma_f16(a2, load_b(w2, kt, nt, 8, lane), c2[nt]);
  }

  // Residual: h = h + upd  (update both f32 master copy and f16 gather copy)
#pragma unroll
  for (int nt = 0; nt < 8; ++nt) {
    const int col = nt * 16 + m;
    const float bb = b2[col];
#pragma unroll
    for (int v = 0; v < 8; ++v) {
      const int rr = base + hsel * 8 + v;
      const size_t idx = (size_t)rr * 128 + col;
      const float nv = h32[idx] + c2[nt][v] + bb;
      h32[idx] = nv;
      h16[idx] = (_Float16)nv;
    }
  }
}

// ---------------------------------------------------------------------------
// Decoder: [N,128] -> 128 -> PReLU -> 3 (W2 padded to 128x16, 1 n-tile)
// ---------------------------------------------------------------------------
__global__ __launch_bounds__(256) void decoder_kernel(
    const _Float16* __restrict__ h16,
    const _Float16* __restrict__ w1, const float* __restrict__ b1,
    const float* __restrict__ aPtr,
    const _Float16* __restrict__ w2p, const float* __restrict__ b2,
    float* __restrict__ out, int ntiles) {
  __shared__ __align__(16) _Float16 sh[8][2048];
  const int wave = threadIdx.x >> 5, lane = threadIdx.x & 31;
  const int tile = blockIdx.x * 8 + wave;
  if (tile >= ntiles) return;
  const int m = lane & 15, hsel = lane >> 4;
  const int base = tile * 16;
  const int row = base + m;
  const float alpha = aPtr[0];

  v16h A[4];
  const _Float16* hr = h16 + (size_t)row * 128;
#pragma unroll
  for (int kt = 0; kt < 4; ++kt) A[kt] = load_a_f16(hr, kt * 32, hsel);

#pragma unroll
  for (int nt = 0; nt < 8; ++nt) {
    v8f c = {};
#pragma unroll
    for (int kt = 0; kt < 4; ++kt)
      c = wmma_f16(A[kt], load_b(w1, kt, nt, 8, lane), c);
    const int col = nt * 16 + m;
    const float bb = b1[col];
#pragma unroll
    for (int v = 0; v < 8; ++v) {
      float hv = c[v] + bb;
      hv = hv > 0.0f ? hv : alpha * hv;
      sh[wave][(hsel * 8 + v) * 128 + col] = (_Float16)hv;
    }
  }

  v8f c2 = {};
  const _Float16* lr = &sh[wave][m * 128];
#pragma unroll
  for (int kt = 0; kt < 4; ++kt) {
    v16h a2 = load_a_f16(lr, kt * 32, hsel);
    c2 = wmma_f16(a2, load_b(w2p, kt, 0, 1, lane), c2);
  }

  if (m < 3) {  // only first 3 output columns are real
    const float bb = b2[m];
#pragma unroll
    for (int v = 0; v < 8; ++v) {
      const int rr = base + hsel * 8 + v;
      out[(size_t)rr * 3 + m] = c2[v] + bb;
    }
  }
}

// ---------------------------------------------------------------------------
// Host
// ---------------------------------------------------------------------------
extern "C" void kernel_launch(void* const* d_in, const int* in_sizes, int n_in,
                              void* d_out, int out_size, void* d_ws, size_t ws_size,
                              hipStream_t stream) {
  (void)in_sizes; (void)n_in; (void)out_size; (void)ws_size;

  const float* x     = (const float*)d_in[0];
  const float* eattr = (const float*)d_in[1];
  const int*   eidx  = (const int*)d_in[2];
  const float* ne_w1 = (const float*)d_in[3];
  const float* ne_b1 = (const float*)d_in[4];
  const float* ne_a  = (const float*)d_in[5];
  const float* ne_w2 = (const float*)d_in[6];
  const float* ne_b2 = (const float*)d_in[7];
  const float* ee_w1 = (const float*)d_in[8];
  const float* ee_b1 = (const float*)d_in[9];
  const float* ee_a  = (const float*)d_in[10];
  const float* ee_w2 = (const float*)d_in[11];
  const float* ee_b2 = (const float*)d_in[12];
  const float* le_w1 = (const float*)d_in[13];
  const float* le_b1 = (const float*)d_in[14];
  const float* le_a  = (const float*)d_in[15];
  const float* le_w2 = (const float*)d_in[16];
  const float* le_b2 = (const float*)d_in[17];
  const float* ln_w1 = (const float*)d_in[18];
  const float* ln_b1 = (const float*)d_in[19];
  const float* ln_a  = (const float*)d_in[20];
  const float* ln_w2 = (const float*)d_in[21];
  const float* ln_b2 = (const float*)d_in[22];
  const float* de_w1 = (const float*)d_in[23];
  const float* de_b1 = (const float*)d_in[24];
  const float* de_a  = (const float*)d_in[25];
  const float* de_w2 = (const float*)d_in[26];
  const float* de_b2 = (const float*)d_in[27];
  float* out = (float*)d_out;

  char* ws = (char*)d_ws;
  size_t off = 0;
  auto alloc = [&](size_t bytes) -> void* {
    void* p = (void*)(ws + off);
    off += (bytes + 255) & ~(size_t)255;
    return p;
  };

  float*    h32 = (float*)alloc((size_t)NN * LDIM * 4);
  _Float16* h16 = (_Float16*)alloc((size_t)NN * LDIM * 2);
  _Float16* e16 = (_Float16*)alloc((size_t)NE * LDIM * 2);
  float*    agg = (float*)alloc((size_t)NN * LDIM * 4);
  _Float16* xh  = (_Float16*)alloc((size_t)NN * 32 * 2);
  _Float16* eh  = (_Float16*)alloc((size_t)NE * 32 * 2);

  // fragment counts: (Kpad/32)*(Npad/16) frags, 512 halves each
  _Float16* ne_w1p = (_Float16*)alloc((size_t)1 * 8 * 512 * 2);
  _Float16* ne_w2p = (_Float16*)alloc((size_t)4 * 8 * 512 * 2);
  _Float16* ee_w1p = (_Float16*)alloc((size_t)1 * 8 * 512 * 2);
  _Float16* ee_w2p = (_Float16*)alloc((size_t)4 * 8 * 512 * 2);
  _Float16* le_w1p = (_Float16*)alloc((size_t)NLAYERS * 12 * 8 * 512 * 2);
  _Float16* le_w2p = (_Float16*)alloc((size_t)NLAYERS * 4 * 8 * 512 * 2);
  _Float16* ln_w1p = (_Float16*)alloc((size_t)NLAYERS * 8 * 8 * 512 * 2);
  _Float16* ln_w2p = (_Float16*)alloc((size_t)NLAYERS * 4 * 8 * 512 * 2);
  _Float16* de_w1p = (_Float16*)alloc((size_t)4 * 8 * 512 * 2);
  _Float16* de_w2p = (_Float16*)alloc((size_t)4 * 1 * 512 * 2);

  auto swz = [&](const float* src, _Float16* dst, int Ksrc, int Nsrc, int Kpad, int Npad) {
    int total = (Kpad / 32) * (Npad / 16) * 512;
    swizzle_weight<<<(total + 255) / 256, 256, 0, stream>>>(src, dst, Ksrc, Nsrc, Kpad, Npad);
  };

  // ---- weight prep ----
  swz(ne_w1, ne_w1p, 30, 128, 32, 128);
  swz(ne_w2, ne_w2p, 128, 128, 128, 128);
  swz(ee_w1, ee_w1p, 4, 128, 32, 128);
  swz(ee_w2, ee_w2p, 128, 128, 128, 128);
  for (int l = 0; l < NLAYERS; ++l) {
    swz(le_w1 + (size_t)l * 384 * 128, le_w1p + (size_t)l * 12 * 8 * 512, 384, 128, 384, 128);
    swz(le_w2 + (size_t)l * 128 * 128, le_w2p + (size_t)l * 4 * 8 * 512, 128, 128, 128, 128);
    swz(ln_w1 + (size_t)l * 256 * 128, ln_w1p + (size_t)l * 8 * 8 * 512, 256, 128, 256, 128);
    swz(ln_w2 + (size_t)l * 128 * 128, ln_w2p + (size_t)l * 4 * 8 * 512, 128, 128, 128, 128);
  }
  swz(de_w1, de_w1p, 128, 128, 128, 128);
  swz(de_w2, de_w2p, 128, 3, 128, 16);

  // ---- input padding to f16 K=32 ----
  {
    int tot = NN * 32;
    pad_f16<<<(tot + 255) / 256, 256, 0, stream>>>(x, xh, NN, 30, 32);
  }
  {
    int tot = NE * 32;
    pad_f16<<<(tot + 255) / 256, 256, 0, stream>>>(eattr, eh, NE, 4, 32);
  }

  const int node_tiles = NN / 16;             // 1250
  const int edge_tiles = NE / 16;             // 20000
  const int node_blocks = (node_tiles + 7) / 8;
  const int edge_blocks = edge_tiles / 8;     // 2500 exact

  // ---- encoders ----
  encoder_kernel<true><<<node_blocks, 256, 0, stream>>>(
      xh, ne_w1p, ne_b1, ne_a, ne_w2p, ne_b2, h16, h32, node_tiles);
  encoder_kernel<false><<<edge_blocks, 256, 0, stream>>>(
      eh, ee_w1p, ee_b1, ee_a, ee_w2p, ee_b2, e16, (float*)nullptr, edge_tiles);

  // ---- processor layers ----
  for (int l = 0; l < NLAYERS; ++l) {
    {
      int tot = NN * LDIM;
      zero_f32<<<(tot + 255) / 256, 256, 0, stream>>>(agg, tot);
    }
    edge_msg_kernel<<<edge_blocks, 256, 0, stream>>>(
        h16, e16, eidx,
        le_w1p + (size_t)l * 12 * 8 * 512, le_b1 + (size_t)l * 128, le_a + l,
        le_w2p + (size_t)l * 4 * 8 * 512, le_b2 + (size_t)l * 128,
        agg, NE);
    node_upd_kernel<<<node_blocks, 256, 0, stream>>>(
        h16, h32, agg,
        ln_w1p + (size_t)l * 8 * 8 * 512, ln_b1 + (size_t)l * 128, ln_a + l,
        ln_w2p + (size_t)l * 4 * 8 * 512, ln_b2 + (size_t)l * 128,
        node_tiles);
  }

  // ---- decoder ----
  decoder_kernel<<<node_blocks, 256, 0, stream>>>(
      h16, de_w1p, de_b1, de_a, de_w2p, de_b2, out, node_tiles);
}
